// DeformCrossAttention2D_9208409883240
// MI455X (gfx1250) — matrix operand order repaired and checked by reading, not compiled, not run on hardware
//
#include <hip/hip_runtime.h>
#include <hip/hip_bf16.h>

// ---------------- shapes ----------------
constexpr int kH    = 50;      // spatial
constexpr int kHW   = 2500;    // H*W
constexpr int kHK   = 12;      // kv grid side
constexpr int kNJ   = 144;     // kv points per head (12*12)
constexpr int kNJP  = 160;     // kNJ padded to WMMA K multiple of 32
constexpr int kDH   = 64;      // dim per head
constexpr int kInner= 512;     // heads * dim_head
constexpr int kNTI  = 157;     // ceil(2500/16) query tiles

// ---------------- workspace layout (bytes) ----------------
constexpr size_t OFF_Q    = 0;                        // q f32        512*2500*4   = 5,120,000
constexpr size_t OFF_QH   = 5120000;                  // q*scale f16  8*2500*64*2  = 2,560,000
constexpr size_t OFF_K    = 7680000;                  // k f16 [h][j][d]           =   147,456
constexpr size_t OFF_VT   = 7827456;                  // v f16 [h][d][160]         =   163,840
constexpr size_t OFF_VG   = 7991296;                  // vgrid f32 [8][144][2]     =     9,216
constexpr size_t OFF_BIAS = 8000512;                  // bias f16 [8][2500][144]   = 5,760,000
constexpr size_t OFF_AO   = 13760512;                 // attn out f16 [2512][512]  = 2,572,288
constexpr size_t OFF_WO   = 16332800;                 // w_out f16 [256][512]      =   262,144

// ---------------- WMMA helpers ----------------
typedef __attribute__((ext_vector_type(16))) _Float16 v16h;
typedef __attribute__((ext_vector_type(8)))  float    v8f;

#define WMMA_F16(a, b, c) \
  __builtin_amdgcn_wmma_f32_16x16x32_f16(false, (a), false, (b), (short)0, (c), false, false)

// A-matrix 16x32 fragment: lane = (hi<<4)|m ; halves 0..7 = K[hi*8 + 0..7],
// halves 8..15 = K[16 + hi*8 + 0..7]. p points at row start + k0 (16B aligned).
__device__ __forceinline__ v16h fragA(const _Float16* p, int hi) {
  v16h f;
  const _Float16* q = p + hi * 8;
  ((uint4*)&f)[0] = *(const uint4*)(q);
  ((uint4*)&f)[1] = *(const uint4*)(q + 16);
  return f;
}
// B-matrix 32x16 fragment: lane = (hi<<4)|n ; halves 0..7 = K[hi*8 + 0..7],
// halves 8..15 = K[16 + hi*8 + 0..7].
__device__ __forceinline__ v16h fragB(const _Float16* p, int hi) {
  v16h f;
  const _Float16* q = p + hi * 8;
  ((uint4*)&f)[0] = *(const uint4*)(q);
  ((uint4*)&f)[1] = *(const uint4*)(q + 16);
  return f;
}

// ---------------- K1: grouped 1x1 q projection ----------------
__global__ void __launch_bounds__(256)
k_qproj(const float* __restrict__ x1, const float* __restrict__ wq,
        float* __restrict__ qf, _Float16* __restrict__ qh) {
  int idx = blockIdx.x * 256 + threadIdx.x;          // 512*2500 exact
  int c = idx / kHW, p = idx % kHW;
  int g = c >> 6;
  const float* xin = x1 + (size_t)g * 32 * kHW + p;
  const float* w = wq + c * 32;
  float acc = 0.f;
#pragma unroll
  for (int ci = 0; ci < 32; ++ci) acc = fmaf(w[ci], xin[ci * kHW], acc);
  qf[(size_t)c * kHW + p] = acc;
  int h = c >> 6, d = c & 63;
  qh[((size_t)h * kHW + p) * kDH + d] = (_Float16)(acc * 0.125f);  // *dim_head^-0.5
}

__global__ void __launch_bounds__(256)
k_cvt_wout(const float* __restrict__ w, _Float16* __restrict__ o) {
  int i = blockIdx.x * 256 + threadIdx.x;            // 256*512 exact
  o[i] = (_Float16)w[i];
}

// ---------------- K2: offsets (depthwise conv + GELU + 1x1 + tanh) ----------------
__global__ void __launch_bounds__(64)
k_offsets(const float* __restrict__ qf, const float* __restrict__ w1,
          const float* __restrict__ b1, const float* __restrict__ w2,
          float* __restrict__ vgrid) {
  __shared__ float s0[64], s1[64];
  int b = blockIdx.x;                  // g*144 + j
  int g = b / kNJ, j = b % kNJ;
  int oy = j / kHK, ox = j % kHK;
  int t = threadIdx.x;                 // channel within group
  const float* img = qf + (size_t)(g * 64 + t) * kHW;
  const float* wk6 = w1 + t * 36;
  float acc = b1[t];
#pragma unroll
  for (int r = 0; r < 6; ++r) {
    int iy = oy * 4 - 1 + r;
    if (iy < 0 || iy >= kH) continue;
#pragma unroll
    for (int s = 0; s < 6; ++s) {
      int ix = ox * 4 - 1 + s;
      if (ix < 0 || ix >= kH) continue;
      acc = fmaf(wk6[r * 6 + s], img[iy * kH + ix], acc);
    }
  }
  float ge = 0.5f * acc * (1.0f + erff(acc * 0.70710678118654752f));  // exact GELU
  s0[t] = ge * w2[t];
  s1[t] = ge * w2[64 + t];
  __syncthreads();
  if (t == 0) {
    float a0 = 0.f, a1 = 0.f;
    for (int i = 0; i < 64; ++i) { a0 += s0[i]; a1 += s1[i]; }
    float offx = tanhf(a0) * 4.0f;
    float offy = tanhf(a1) * 4.0f;
    vgrid[(size_t)b * 2 + 0] = 2.0f * ((float)ox + offx) / (float)(kHK - 1) - 1.0f;
    vgrid[(size_t)b * 2 + 1] = 2.0f * ((float)oy + offy) / (float)(kHK - 1) - 1.0f;
  }
}

// ---------------- K3: grid-sample + grouped k/v projections ----------------
__global__ void __launch_bounds__(256)
k_kv(const float* __restrict__ x2, const float* __restrict__ vgrid,
     const float* __restrict__ wk, const float* __restrict__ wv,
     _Float16* __restrict__ k16, _Float16* __restrict__ vT16) {
  __shared__ float kv[32 * kNJ];
  int g = blockIdx.x;
  int tid = threadIdx.x;
  for (int e = tid; e < 32 * kNJ; e += 256) {
    int ci = e / kNJ, j = e % kNJ;
    float gx = vgrid[(g * kNJ + j) * 2 + 0];
    float gy = vgrid[(g * kNJ + j) * 2 + 1];
    float ix = ((gx + 1.0f) * (float)kH - 1.0f) * 0.5f;
    float iy = ((gy + 1.0f) * (float)kH - 1.0f) * 0.5f;
    float x0f = floorf(ix), y0f = floorf(iy);
    float wx = ix - x0f, wy = iy - y0f;
    int x0 = (int)x0f, y0 = (int)y0f;
    const float* img = x2 + (size_t)(g * 32 + ci) * kHW;
    float v = 0.f;
#pragma unroll
    for (int dy = 0; dy < 2; ++dy)
#pragma unroll
      for (int dx = 0; dx < 2; ++dx) {
        int xi = x0 + dx, yi = y0 + dy;
        float wgt = (dx ? wx : 1.f - wx) * (dy ? wy : 1.f - wy);
        if (xi >= 0 && xi < kH && yi >= 0 && yi < kH)
          v = fmaf(wgt, img[yi * kH + xi], v);
      }
    kv[ci * kNJ + j] = v;
  }
  __syncthreads();
  for (int o = tid; o < 64 * kNJ; o += 256) {
    int d = o / kNJ, j = o % kNJ;
    const float* wkr = wk + (size_t)(g * 64 + d) * 32;
    const float* wvr = wv + (size_t)(g * 64 + d) * 32;
    float ck = 0.f, cv = 0.f;
#pragma unroll
    for (int ci = 0; ci < 32; ++ci) {
      float kvv = kv[ci * kNJ + j];
      ck = fmaf(wkr[ci], kvv, ck);
      cv = fmaf(wvr[ci], kvv, cv);
    }
    k16[((size_t)g * kNJ + j) * kDH + d] = (_Float16)ck;     // [h][j][d]
    vT16[((size_t)g * kDH + d) * kNJP + j] = (_Float16)cv;   // [h][d][j]
  }
  for (int e = tid; e < 64 * 16; e += 256) {                 // zero K-pad rows 144..159
    int d = e / 16, j = kNJ + (e % 16);
    vT16[((size_t)g * kDH + d) * kNJP + j] = (_Float16)0.f;
  }
}

// ---------------- K4: CPB bias MLP (WMMA layer 2) ----------------
__global__ void __launch_bounds__(256)
k_bias(const float* __restrict__ vgrid, const float* __restrict__ w1,
       const float* __restrict__ b1, const float* __restrict__ w2,
       const float* __restrict__ b2, const float* __restrict__ w3,
       const float* __restrict__ b3, _Float16* __restrict__ bias16) {
  __shared__ __align__(16) _Float16 H1[256][64];
  __shared__ __align__(16) _Float16 W2s[64][64];
  __shared__ float bacc[256];
  int g  = blockIdx.y;
  int it = blockIdx.x / 9, jt = blockIdx.x % 9;
  int tid = threadIdx.x;
  for (int e = tid; e < 64 * 64; e += 256) ((_Float16*)W2s)[e] = (_Float16)w2[e];
  bacc[tid] = 0.f;
  {  // layer 1 (2 -> 64) + relu, one (i,j) pair per thread
    int ii = tid >> 4, jj = tid & 15;
    int i = min(it * 16 + ii, kHW - 1);
    int j = jt * 16 + jj;
    float qx = 2.0f * (float)(i % kH) / (float)(kH - 1) - 1.0f;
    float qy = 2.0f * (float)(i / kH) / (float)(kH - 1) - 1.0f;
    float p0 = qx - vgrid[(g * kNJ + j) * 2 + 0];
    float p1 = qy - vgrid[(g * kNJ + j) * 2 + 1];
    float s0 = copysignf(log1pf(fabsf(p0)), p0);
    float s1 = copysignf(log1pf(fabsf(p1)), p1);
#pragma unroll
    for (int f = 0; f < 64; ++f) {
      float h = fmaf(w1[f * 2], s0, fmaf(w1[f * 2 + 1], s1, b1[f]));
      H1[tid][f] = (_Float16)fmaxf(h, 0.f);
    }
  }
  __syncthreads();
  int wav = tid >> 5, lane = tid & 31;
  int hi = lane >> 4, ml = lane & 15;
#pragma unroll
  for (int mi = 0; mi < 2; ++mi) {     // layer 2 (64 -> 64) WMMA + relu, layer 3 dot
    int mt = wav * 2 + mi;
    const _Float16* arow = &H1[mt * 16 + ml][0];
    v16h a0 = fragA(arow, hi);
    v16h a1 = fragA(arow + 32, hi);
    float part[8] = {};
#pragma unroll
    for (int nt = 0; nt < 4; ++nt) {
      int n = nt * 16 + ml;
      const _Float16* brow = &W2s[n][0];
      v8f c = {};
      c = WMMA_F16(a0, fragB(brow, hi), c);
      c = WMMA_F16(a1, fragB(brow + 32, hi), c);
      float bb = b2[n], ww = w3[n];
#pragma unroll
      for (int r = 0; r < 8; ++r)
        part[r] = fmaf(fmaxf(c[r] + bb, 0.f), ww, part[r]);
    }
#pragma unroll
    for (int r = 0; r < 8; ++r)
      atomicAdd(&bacc[mt * 16 + hi * 8 + r], part[r]);
  }
  __syncthreads();
  {
    int ii = tid >> 4, jj = tid & 15;
    int i = it * 16 + ii;
    if (i < kHW) {
      int j = jt * 16 + jj;
      bias16[((size_t)g * kHW + i) * kNJ + j] = (_Float16)(bacc[tid] + b3[0]);
    }
  }
}

// ---------------- K5: fused attention (QK^T -> +bias -> softmax -> PV) ----------------
__global__ void __launch_bounds__(128)
k_attn(const _Float16* __restrict__ qh, const _Float16* __restrict__ k16,
       const _Float16* __restrict__ vT16, const _Float16* __restrict__ bias16,
       _Float16* __restrict__ ao16) {
  __shared__ __align__(16) float    S[4][16][kNJ];
  __shared__ __align__(16) _Float16 P[4][16][kNJP];
  int h = blockIdx.y;
  int tid = threadIdx.x;
  int w = tid >> 5, lane = tid & 31;
  int hi = lane >> 4, ml = lane & 15;
  int tile = min((int)blockIdx.x * 4 + w, kNTI - 1);
  int i0 = tile * 16;
  {  // sim = (q*scale) @ k^T  +  bias
    int iA = min(i0 + ml, kHW - 1);
    const _Float16* qrow = qh + ((size_t)h * kHW + iA) * kDH;
    v16h a0 = fragA(qrow, hi);
    v16h a1 = fragA(qrow + 32, hi);
#pragma unroll
    for (int nt = 0; nt < 9; ++nt) {
      const _Float16* krow = k16 + ((size_t)h * kNJ + nt * 16 + ml) * kDH;
      v8f c = {};
      c = WMMA_F16(a0, fragB(krow, hi), c);
      c = WMMA_F16(a1, fragB(krow + 32, hi), c);
      int n = nt * 16 + ml;
#pragma unroll
      for (int r = 0; r < 8; ++r) {
        int m = hi * 8 + r;
        int ib = min(i0 + m, kHW - 1);
        float bb = (float)bias16[((size_t)h * kHW + ib) * kNJ + n];
        S[w][m][n] = c[r] + bb;
      }
    }
  }
  __syncthreads();
  {  // softmax over 144, two lanes per row
    int m = lane >> 1, half = lane & 1, j0 = half * 72;
    float mx = -3.0e38f;
    for (int jj = 0; jj < 72; ++jj) mx = fmaxf(mx, S[w][m][j0 + jj]);
    mx = fmaxf(mx, __shfl_xor(mx, 1));
    float sum = 0.f;
    for (int jj = 0; jj < 72; ++jj) {
      float e = __expf(S[w][m][j0 + jj] - mx);
      S[w][m][j0 + jj] = e;
      sum += e;
    }
    sum += __shfl_xor(sum, 1);
    float inv = 1.0f / sum;
    for (int jj = 0; jj < 72; ++jj)
      P[w][m][j0 + jj] = (_Float16)(S[w][m][j0 + jj] * inv);
#pragma unroll
    for (int jj = 0; jj < 8; ++jj)       // zero the K-pad 144..159
      P[w][m][kNJ + half * 8 + jj] = (_Float16)0.f;
  }
  __syncthreads();
  {  // out = P @ v   (K padded to 160)
    v8f o0 = {}, o1 = {}, o2 = {}, o3 = {};
    const _Float16* prow = &P[w][ml][0];
#pragma unroll
    for (int kk = 0; kk < 5; ++kk) {
      v16h a = fragA(prow + kk * 32, hi);
      const _Float16* vb = vT16 + (size_t)h * kDH * kNJP + kk * 32;
      o0 = WMMA_F16(a, fragB(vb + (size_t)(0 * 16 + ml) * kNJP, hi), o0);
      o1 = WMMA_F16(a, fragB(vb + (size_t)(1 * 16 + ml) * kNJP, hi), o1);
      o2 = WMMA_F16(a, fragB(vb + (size_t)(2 * 16 + ml) * kNJP, hi), o2);
      o3 = WMMA_F16(a, fragB(vb + (size_t)(3 * 16 + ml) * kNJP, hi), o3);
    }
    v8f oo[4] = {o0, o1, o2, o3};
#pragma unroll
    for (int nt = 0; nt < 4; ++nt) {
      int d = nt * 16 + ml;
#pragma unroll
      for (int r = 0; r < 8; ++r) {
        int i = i0 + hi * 8 + r;
        if (i < kHW)
          ao16[(size_t)i * kInner + h * kDH + d] = (_Float16)oo[nt][r];
      }
    }
  }
}

// ---------------- K6: output projection (2500x512)@(512x256) + bias ----------------
__global__ void __launch_bounds__(128)
k_outproj(const _Float16* __restrict__ ao16, const _Float16* __restrict__ wo16,
          const float* __restrict__ b_out, float* __restrict__ out) {
  int tid = threadIdx.x;
  int w = tid >> 5, lane = tid & 31;
  int hi = lane >> 4, ml = lane & 15;
  int tile = min((int)blockIdx.x * 4 + w, kNTI - 1);
  int i0 = tile * 16;
  int iA = min(i0 + ml, kHW - 1);
  const _Float16* arow = ao16 + (size_t)iA * kInner;
  __builtin_prefetch(arow, 0, 3);
  for (int nt = 0; nt < 16; ++nt) {
    int n = nt * 16 + ml;
    const _Float16* brow = wo16 + (size_t)n * kInner;
    v8f c = {};
#pragma unroll
    for (int ks = 0; ks < 16; ++ks)
      c = WMMA_F16(fragA(arow + ks * 32, hi), fragB(brow + ks * 32, hi), c);
    float bo = b_out[n];
#pragma unroll
    for (int r = 0; r < 8; ++r) {
      int i = i0 + hi * 8 + r;
      if (i < kHW) out[(size_t)n * kHW + i] = c[r] + bo;
    }
  }
}

// ---------------- launch ----------------
extern "C" void kernel_launch(void* const* d_in, const int* in_sizes, int n_in,
                              void* d_out, int out_size, void* d_ws, size_t ws_size,
                              hipStream_t stream) {
  const float* x1     = (const float*)d_in[0];
  const float* x2     = (const float*)d_in[1];
  const float* wq     = (const float*)d_in[2];
  const float* wk     = (const float*)d_in[3];
  const float* wv     = (const float*)d_in[4];
  const float* w_out  = (const float*)d_in[5];
  const float* b_out  = (const float*)d_in[6];
  const float* off_w1 = (const float*)d_in[7];
  const float* off_b1 = (const float*)d_in[8];
  const float* off_w2 = (const float*)d_in[9];
  const float* cpb_w1 = (const float*)d_in[10];
  const float* cpb_b1 = (const float*)d_in[11];
  const float* cpb_w2 = (const float*)d_in[12];
  const float* cpb_b2 = (const float*)d_in[13];
  const float* cpb_w3 = (const float*)d_in[14];
  const float* cpb_b3 = (const float*)d_in[15];
  (void)in_sizes; (void)n_in; (void)out_size; (void)ws_size;

  char* ws = (char*)d_ws;
  float*    qf     = (float*)(ws + OFF_Q);
  _Float16* qh     = (_Float16*)(ws + OFF_QH);
  _Float16* k16    = (_Float16*)(ws + OFF_K);
  _Float16* vT16   = (_Float16*)(ws + OFF_VT);
  float*    vgrid  = (float*)(ws + OFF_VG);
  _Float16* bias16 = (_Float16*)(ws + OFF_BIAS);
  _Float16* ao16   = (_Float16*)(ws + OFF_AO);
  _Float16* wo16   = (_Float16*)(ws + OFF_WO);
  float*    out    = (float*)d_out;

  k_qproj   <<<5000, 256, 0, stream>>>(x1, wq, qf, qh);
  k_cvt_wout<<<512, 256, 0, stream>>>(w_out, wo16);
  k_offsets <<<8 * kNJ, 64, 0, stream>>>(qf, off_w1, off_b1, off_w2, vgrid);
  k_kv      <<<8, 256, 0, stream>>>(x2, vgrid, wk, wv, k16, vT16);
  k_bias    <<<dim3(kNTI * 9, 8), 256, 0, stream>>>(vgrid, cpb_w1, cpb_b1, cpb_w2,
                                                    cpb_b2, cpb_w3, cpb_b3, bias16);
  k_attn    <<<dim3(40, 8), 128, 0, stream>>>(qh, k16, vT16, bias16, ao16);
  k_outproj <<<40, 128, 0, stream>>>(ao16, wo16, b_out, out);
}
